// MultiHeadAttention_65343632441706
// MI455X (gfx1250) — matrix operand order repaired and checked by reading, not compiled, not run on hardware
//
#include <hip/hip_runtime.h>
#include <hip/hip_bf16.h>
#include <cstddef>
#include <cstdint>

typedef __attribute__((ext_vector_type(16))) _Float16 v16h;
typedef __attribute__((ext_vector_type(8)))  float    v8f;

static constexpr int B      = 8;
static constexpr int L      = 1024;
static constexpr int DM     = 1024;   // d_model
static constexpr int H      = 16;     // heads
static constexpr int DK     = 64;     // head dim
static constexpr float LN_EPS  = 1e-5f;
static constexpr float NEG_INF = -1e9f;
static constexpr int M_ROWS = B * L;  // 8192

// Detect gfx1250 async global->LDS builtins (guarded; signature per hipcc
// diagnostic: param0 is a 16-byte int-vector pointer).
#if defined(__has_builtin)
#if __has_builtin(__builtin_amdgcn_global_load_async_to_lds_b128)
#define HAVE_ASYNC_LDS 1
#endif
#endif
#ifndef HAVE_ASYNC_LDS
#define HAVE_ASYNC_LDS 0
#endif

#if HAVE_ASYNC_LDS
typedef int v4i __attribute__((vector_size(16)));
typedef __attribute__((address_space(3))) v4i lds_v4i;
__device__ __forceinline__ void wait_asynccnt0() {
#if __has_builtin(__builtin_amdgcn_s_wait_asynccnt)
    __builtin_amdgcn_s_wait_asynccnt(0);
#else
    asm volatile("s_wait_asynccnt 0" ::: "memory");
#endif
}
#endif

// ---------------------------------------------------------------------------
// CDNA5 WMMA 16x16x32 f16 fragment index helpers (wave32; contiguous k order used by this code)
// A (16x32, f16): lane holds row M=lane&15. Element e (0..15):
//   K = ((lane&16)?16:0) + e
// B (32x16, f16): lane holds col N=lane&15. Element e: K = ((lane&16)?16:0)+e
// C/D (16x16, f32): vgpr r holds M = r + ((lane&16)?8:0), N = lane&15
// ---------------------------------------------------------------------------
__device__ __forceinline__ int a_frag_k(int lane, int e) {
    return ((lane & 16) ? 16 : 0) + e;
}

// ===========================================================================
// fp32 -> f16 conversion (one-time, halves all GEMM streaming traffic)
// ===========================================================================
__global__ __launch_bounds__(256)
void cvt_kernel(const float* __restrict__ src, _Float16* __restrict__ dst, int n4)
{
    int i = blockIdx.x * 256 + threadIdx.x;
    if (i < n4) {
        float4 v = ((const float4*)src)[i];
        dst[i * 4 + 0] = (_Float16)v.x;
        dst[i * 4 + 1] = (_Float16)v.y;
        dst[i * 4 + 2] = (_Float16)v.z;
        dst[i * 4 + 3] = (_Float16)v.w;
    }
}

// ===========================================================================
// Tiled GEMM: C[M_ROWS x DM] = A16[M_ROWS x DM] @ W16[DM x DM] + bias (+res)
// Block tile 128x64, 256 threads = 8 waves (4 M-groups x 2 N-groups),
// each wave -> 32x32 output (2x2 accumulators, 4 WMMA / K-step of 32).
// A tile staged via async global->LDS DMA when available.
// OUT_MODE 0: f16 [B,H,L,DK]   OUT_MODE 1: f16 [B,H,DK,L] (V^T)
// OUT_MODE 2: f32 [row,col] = acc + bias + residual (pre-LayerNorm)
// ===========================================================================
template <int OUT_MODE>
__global__ __launch_bounds__(256)
void gemm_kernel(const _Float16* __restrict__ A, const _Float16* __restrict__ W,
                 const float* __restrict__ bias, const float* __restrict__ residual,
                 void* __restrict__ out)
{
    __shared__ _Float16 ldsA[128 * 32];  // [m][k]
    __shared__ _Float16 ldsB[64 * 32];   // [n][k] (W tile transposed)

    const int tid  = threadIdx.x;
    const int lane = tid & 31;
    const int w    = tid >> 5;
    const int wmRow = (w & 3) * 32;      // this wave's 32 M rows
    const int wnCol = (w >> 2) * 32;     // this wave's 32 N cols
    const int bn   = blockIdx.x * 64;
    const int bm   = blockIdx.y * 128;

    v8f c[2][2] = {};

    for (int kk = 0; kk < DM; kk += 32) {
        // ---- stage A tile 128x32 f16 (8 KB) ----
#if HAVE_ASYNC_LDS
        {
            #pragma unroll
            for (int i = 0; i < 2; ++i) {
                int cidx = tid * 2 + i;          // 512 x 16-byte chunks
                int m  = cidx >> 2;
                int ko = (cidx & 3) * 8;
                __builtin_amdgcn_global_load_async_to_lds_b128(
                    (v4i*)&A[(size_t)(bm + m) * DM + kk + ko],
                    (lds_v4i*)&ldsA[m * 32 + ko], 0, 0);
            }
        }
#else
        #pragma unroll
        for (int i = 0; i < 16; ++i) {
            int idx = tid * 16 + i;
            int m = idx >> 5, k = idx & 31;
            ldsA[idx] = A[(size_t)(bm + m) * DM + kk + k];
        }
#endif
        // ---- stage W tile transposed -> [n][k] (4 KB) ----
        #pragma unroll
        for (int i = 0; i < 8; ++i) {
            int idx = tid * 8 + i;
            int k = idx >> 6, n = idx & 63;
            ldsB[n * 32 + k] = W[(size_t)(kk + k) * DM + bn + n];
        }
#if HAVE_ASYNC_LDS
        wait_asynccnt0();
#endif
        __syncthreads();

        // ---- fragments + 4 WMMA ----
        v16h a[2], b[2];
        #pragma unroll
        for (int e = 0; e < 16; ++e) {
            int ak = a_frag_k(lane, e);
            int bk = ((lane & 16) ? 16 : 0) + e;
            a[0][e] = ldsA[(wmRow + (lane & 15)) * 32 + ak];
            a[1][e] = ldsA[(wmRow + 16 + (lane & 15)) * 32 + ak];
            b[0][e] = ldsB[(wnCol + (lane & 15)) * 32 + bk];
            b[1][e] = ldsB[(wnCol + 16 + (lane & 15)) * 32 + bk];
        }
        c[0][0] = __builtin_amdgcn_wmma_f32_16x16x32_f16(false, a[0], false, b[0], (short)0, c[0][0], false, false);
        c[0][1] = __builtin_amdgcn_wmma_f32_16x16x32_f16(false, a[0], false, b[1], (short)0, c[0][1], false, false);
        c[1][0] = __builtin_amdgcn_wmma_f32_16x16x32_f16(false, a[1], false, b[0], (short)0, c[1][0], false, false);
        c[1][1] = __builtin_amdgcn_wmma_f32_16x16x32_f16(false, a[1], false, b[1], (short)0, c[1][1], false, false);
        __syncthreads();
    }

    // ---- epilogue ----
    #pragma unroll
    for (int mi = 0; mi < 2; ++mi) {
        #pragma unroll
        for (int nj = 0; nj < 2; ++nj) {
            int ncol = bn + wnCol + nj * 16 + (lane & 15);
            float bv = bias[ncol];
            #pragma unroll
            for (int r = 0; r < 8; ++r) {
                int row = bm + wmRow + mi * 16 + r + ((lane & 16) ? 8 : 0);
                float val = c[mi][nj][r] + bv;
                int bi = row >> 10, lq = row & 1023;
                int hh = ncol >> 6, d = ncol & 63;
                if (OUT_MODE == 0) {
                    ((_Float16*)out)[((size_t)(bi * H + hh) * L + lq) * DK + d] = (_Float16)val;
                } else if (OUT_MODE == 1) {
                    ((_Float16*)out)[((size_t)(bi * H + hh) * DK + d) * L + lq] = (_Float16)val;
                } else {
                    ((float*)out)[(size_t)row * DM + ncol] =
                        val + residual[(size_t)row * DM + ncol];
                }
            }
        }
    }
}

// ===========================================================================
// Attention: one block = (b, h, 32 query rows). 256 threads = 8 waves.
// LDS: S[32][1024] f32 (128 KB dynamic).
// ===========================================================================
__global__ __launch_bounds__(256)
void attn_kernel(const _Float16* __restrict__ qh, const _Float16* __restrict__ kh,
                 const _Float16* __restrict__ vt, const unsigned char* __restrict__ mask,
                 float* __restrict__ attn_out, _Float16* __restrict__ ctx)
{
    extern __shared__ float S[];   // [32][1024]

    const int tid  = threadIdx.x;
    const int lane = tid & 31;
    const int w    = tid >> 5;
    const int bid  = blockIdx.x;
    const int qblk = bid & 31;
    const int h    = (bid >> 5) & 15;
    const int b    = bid >> 9;
    const int qbase = qblk * 32;
    const size_t headQK = (size_t)(b * H + h) * L * DK;
    const size_t headV  = (size_t)(b * H + h) * DK * L;

    // ---- Phase 1: S = (Q K^T)/8, mask ----
    {
        const int qt = w & 1;
        v16h aq0, aq1;
        #pragma unroll
        for (int e = 0; e < 16; ++e) {
            int m = lane & 15;
            size_t rowoff = headQK + (size_t)(qbase + qt * 16 + m) * DK;
            aq0[e] = qh[rowoff + a_frag_k(lane, e)];
            aq1[e] = qh[rowoff + 32 + a_frag_k(lane, e)];
        }
        for (int kt = (w >> 1); kt < 64; kt += 4) {
            v8f acc = {};
            int key0 = kt * 16 + (lane & 15);
            const v16h bk0 = *(const v16h*)&kh[headQK + (size_t)key0 * DK + ((lane & 16) ? 16 : 0)];
            const v16h bk1 = *(const v16h*)&kh[headQK + (size_t)key0 * DK + 32 + ((lane & 16) ? 16 : 0)];
            acc = __builtin_amdgcn_wmma_f32_16x16x32_f16(false, aq0, false, bk0, (short)0, acc, false, false);
            acc = __builtin_amdgcn_wmma_f32_16x16x32_f16(false, aq1, false, bk1, (short)0, acc, false, false);
            #pragma unroll
            for (int r = 0; r < 8; ++r) {
                int qrow = qt * 16 + r + ((lane & 16) ? 8 : 0);
                int key  = kt * 16 + (lane & 15);
                float sc = acc[r] * 0.125f;
                if (mask[((size_t)b * L + qbase + qrow) * L + key]) sc = NEG_INF;
                S[qrow * 1024 + key] = sc;
            }
        }
    }
    __syncthreads();

    // ---- Phase 2: softmax (wave-parallel rows) ----
    for (int row = w; row < 32; row += 8) {
        float* Sr = &S[row * 1024];
        float m = -3.0e38f;
        for (int cc = lane; cc < 1024; cc += 32) m = fmaxf(m, Sr[cc]);
        #pragma unroll
        for (int off = 16; off > 0; off >>= 1) m = fmaxf(m, __shfl_xor(m, off, 32));
        float sum = 0.f;
        for (int cc = lane; cc < 1024; cc += 32) {
            float e = __expf(Sr[cc] - m);
            Sr[cc] = e;
            sum += e;
        }
        #pragma unroll
        for (int off = 16; off > 0; off >>= 1) sum += __shfl_xor(sum, off, 32);
        float inv = 1.0f / sum;
        for (int cc = lane; cc < 1024; cc += 32) Sr[cc] *= inv;
    }
    __syncthreads();

    // ---- Phase 3: write attn probabilities (layout [H*B, L, L]) ----
    {
        float* dst = &attn_out[((size_t)(h * B + b) * L + qbase) * L];
        for (int i = tid; i < 32 * 1024; i += 256) dst[i] = S[i];
    }

    // ---- Phase 4: ctx = P @ V ----
    {
        const int qt = w >> 2;
        const int dt = w & 3;
        v8f acc = {};
        for (int kc = 0; kc < 32; ++kc) {
            v16h a;
            #pragma unroll
            for (int e = 0; e < 16; ++e) {
                int k = kc * 32 + a_frag_k(lane, e);
                a[e] = (_Float16)S[(qt * 16 + (lane & 15)) * 1024 + k];
            }
            const v16h bv = *(const v16h*)&vt[headV + (size_t)(dt * 16 + (lane & 15)) * L
                                             + kc * 32 + ((lane & 16) ? 16 : 0)];
            acc = __builtin_amdgcn_wmma_f32_16x16x32_f16(false, a, false, bv, (short)0, acc, false, false);
        }
        #pragma unroll
        for (int r = 0; r < 8; ++r) {
            int qrow = qt * 16 + r + ((lane & 16) ? 8 : 0);
            int d    = dt * 16 + (lane & 15);
            ctx[((size_t)b * L + qbase + qrow) * DM + h * DK + d] = (_Float16)acc[r];
        }
    }
}

// ===========================================================================
// LayerNorm: one block per row of [8192 x 1024]
// ===========================================================================
__global__ __launch_bounds__(256)
void ln_kernel(const float* __restrict__ x, const float* __restrict__ gamma,
               const float* __restrict__ beta, float* __restrict__ out)
{
    __shared__ float rs[256], rq[256];
    const int row = blockIdx.x;
    const int tid = threadIdx.x;
    const float* xr = &x[(size_t)row * DM];

    float s = 0.f, sq = 0.f;
    for (int i = tid; i < DM; i += 256) {
        float v = xr[i];
        s += v; sq += v * v;
    }
    rs[tid] = s; rq[tid] = sq;
    __syncthreads();
    for (int st = 128; st > 0; st >>= 1) {
        if (tid < st) { rs[tid] += rs[tid + st]; rq[tid] += rq[tid + st]; }
        __syncthreads();
    }
    float mean = rs[0] * (1.0f / DM);
    float var  = rq[0] * (1.0f / DM) - mean * mean;
    float inv  = rsqrtf(var + LN_EPS);
    for (int i = tid; i < DM; i += 256)
        out[(size_t)row * DM + i] = (xr[i] - mean) * inv * gamma[i] + beta[i];
}

// ===========================================================================
extern "C" void kernel_launch(void* const* d_in, const int* in_sizes, int n_in,
                              void* d_out, int out_size, void* d_ws, size_t ws_size,
                              hipStream_t stream)
{
    (void)in_sizes; (void)n_in; (void)out_size; (void)ws_size;

    const float* q    = (const float*)d_in[0];
    const float* k    = (const float*)d_in[1];
    const float* v    = (const float*)d_in[2];
    const unsigned char* mask = (const unsigned char*)d_in[3];
    const float* wq   = (const float*)d_in[4];
    const float* bq   = (const float*)d_in[5];
    const float* wk   = (const float*)d_in[6];
    const float* bk   = (const float*)d_in[7];
    const float* wv   = (const float*)d_in[8];
    const float* bv   = (const float*)d_in[9];
    const float* wo   = (const float*)d_in[10];
    const float* bo   = (const float*)d_in[11];
    const float* gamma= (const float*)d_in[12];
    const float* beta = (const float*)d_in[13];

    // workspace carve-up (f16 elements unless noted)
    const size_t E  = (size_t)M_ROWS * DM;   // 8,388,608 elems
    const size_t WE = (size_t)DM * DM;       // 1,048,576 elems
    _Float16* qh   = (_Float16*)d_ws;
    _Float16* kh   = qh  + E;
    _Float16* vt   = kh  + E;
    _Float16* ctx  = vt  + E;
    _Float16* q16  = ctx + E;
    _Float16* k16  = q16 + E;
    _Float16* v16  = k16 + E;
    _Float16* wq16 = v16 + E;
    _Float16* wk16 = wq16 + WE;
    _Float16* wv16 = wk16 + WE;
    _Float16* wo16 = wv16 + WE;
    float*    tmp  = (float*)q16;            // aliases q16+k16 (dead by then)

    float* out_ln   = (float*)d_out;                 // [8192,1024]
    float* out_attn = out_ln + (size_t)M_ROWS * DM;  // [128,1024,1024]

    dim3 blk(256);
    // one-time fp32 -> f16 conversions
    {
        int n4a = (int)(E / 4), n4w = (int)(WE / 4);
        cvt_kernel<<<dim3((n4a + 255) / 256), blk, 0, stream>>>(q,  q16,  n4a);
        cvt_kernel<<<dim3((n4a + 255) / 256), blk, 0, stream>>>(k,  k16,  n4a);
        cvt_kernel<<<dim3((n4a + 255) / 256), blk, 0, stream>>>(v,  v16,  n4a);
        cvt_kernel<<<dim3((n4w + 255) / 256), blk, 0, stream>>>(wq, wq16, n4w);
        cvt_kernel<<<dim3((n4w + 255) / 256), blk, 0, stream>>>(wk, wk16, n4w);
        cvt_kernel<<<dim3((n4w + 255) / 256), blk, 0, stream>>>(wv, wv16, n4w);
        cvt_kernel<<<dim3((n4w + 255) / 256), blk, 0, stream>>>(wo, wo16, n4w);
    }

    dim3 gGrid(DM / 64, M_ROWS / 128);               // (16, 64)

    // Q/K/V projections (f16 in, f16 head-layout out)
    gemm_kernel<0><<<gGrid, blk, 0, stream>>>(q16, wq16, bq, nullptr, qh);
    gemm_kernel<0><<<gGrid, blk, 0, stream>>>(k16, wk16, bk, nullptr, kh);
    gemm_kernel<1><<<gGrid, blk, 0, stream>>>(v16, wv16, bv, nullptr, vt);

    // attention: B*H*(L/32) = 4096 blocks, 128 KB dynamic LDS
    attn_kernel<<<dim3(B * H * (L / 32)), blk, 32 * 1024 * sizeof(float), stream>>>(
        qh, kh, vt, mask, out_attn, ctx);

    // out projection + bias + residual
    gemm_kernel<2><<<gGrid, blk, 0, stream>>>(ctx, wo16, bo, q, tmp);

    // LayerNorm -> final output
    ln_kernel<<<dim3(M_ROWS), blk, 0, stream>>>(tmp, gamma, beta, out_ln);
}
